// PhysicsNetwork_71837622993315
// MI455X (gfx1250) — matrix operand order repaired and checked by reading, not compiled, not run on hardware
//
#include <hip/hip_runtime.h>

typedef __attribute__((ext_vector_type(16))) __bf16 v16bf;
typedef __attribute__((ext_vector_type(8)))  float  v8f;

#define DEV __device__ __forceinline__

DEV unsigned short f2bf(float f) {
  unsigned int u = __float_as_uint(f);
  u += 0x7FFFu + ((u >> 16) & 1u);          // round-to-nearest-even
  return (unsigned short)(u >> 16);
}
DEV float bf2f(unsigned short h) { return __uint_as_float(((unsigned int)h) << 16); }
DEV __bf16 us2b(unsigned short h) { union { unsigned short u; __bf16 b; } c; c.u = h; return c.b; }
DEV float act_elu(float x) { return x > 0.f ? x : (__expf(x) - 1.f); }
DEV float act_sig(float x) { return 1.f / (1.f + __expf(-x)); }

DEV v8f wmma_bf16(v16bf a, v16bf b, v8f c) {
  return __builtin_amdgcn_wmma_f32_16x16x32_bf16(false, a, false, b, (short)0, c, false, false);
}

template <int NT>
DEV void zero_acc(v8f (&a)[NT]) {
#pragma unroll
  for (int t = 0; t < NT; ++t)
#pragma unroll
    for (int e = 0; e < 8; ++e) a[t][e] = 0.f;
}

// A fragment: 16x32 bf16 tile from LDS (row-major, leading dim ld).
// lanes 0-15: M=lane, K=[0..7] + [16..23]; lanes 16-31: M=lane-16, K=[8..15] + [24..31]
DEV v16bf load_a_frag(const unsigned short* sA, int row_base, int k_base, int ld) {
  const int lane = threadIdx.x & 31;
  const unsigned short* p = sA + (row_base + (lane & 15)) * ld + k_base + ((lane >> 4) << 3);
  v16bf a;
#pragma unroll
  for (int h = 0; h < 8; ++h) { a[h] = us2b(p[h]); a[h + 8] = us2b(p[16 + h]); }
  return a;
}

// B fragment from K-transposed LDS chunk sBt[n*32 + k] (32 x NP tile).
// lanes 0-15: N=lane, K=[0..7] + [16..23]; lanes 16-31: N=lane-16, K=[8..15] + [24..31]
DEV v16bf load_b_frag(const unsigned short* sBt, int col_base) {
  const int lane = threadIdx.x & 31;
  const unsigned short* p = sBt + (col_base + (lane & 15)) * 32 + ((lane >> 4) << 3);
  v16bf b;
#pragma unroll
  for (int h = 0; h < 8; ++h) { b[h] = us2b(p[h]); b[h + 8] = us2b(p[16 + h]); }
  return b;
}

// GEMM with A panel resident in LDS; W (KTOT x NREAL fp32, row-major) streamed
// through a K-transposed LDS chunk in K=32 steps. acc: NP/16 tiles of 16x16.
template <int KTOT, int NP, int NREAL, int NTHREADS>
DEV void lds_gemm(const unsigned short* sAp, int a_ld, int a_row,
                  const float* __restrict__ W, unsigned short* sBt, v8f* acc) {
  const int tid = threadIdx.x;
  for (int kc = 0; kc < KTOT; kc += 32) {
    for (int idx = tid; idx < 32 * NP; idx += NTHREADS) {
      const int n = idx >> 5, k = idx & 31;
      float v = (NP == NREAL || n < NREAL) ? W[(long)(kc + k) * NREAL + n] : 0.f;
      sBt[idx] = f2bf(v);
    }
    __syncthreads();
    const v16bf a = load_a_frag(sAp, a_row, kc, a_ld);
#pragma unroll
    for (int t = 0; t < NP / 16; ++t)
      acc[t] = wmma_bf16(a, load_b_frag(sBt, t * 16), acc[t]);
    __syncthreads();
  }
}

// bias + ELU epilogue into LDS, bf16, following C-matrix layout
template <int NP>
DEV void epi_elu_to_lds(const v8f* acc, const float* __restrict__ bias,
                        unsigned short* sOut, int out_ld, int wave) {
  const int lane = threadIdx.x & 31;
#pragma unroll
  for (int t = 0; t < NP / 16; ++t) {
    const int col = t * 16 + (lane & 15);
    const int rb = wave * 16 + ((lane >> 4) << 3);
    const float b = bias[col];
#pragma unroll
    for (int v = 0; v < 8; ++v)
      sOut[(rb + v) * out_ld + col] = f2bf(act_elu(acc[t][v] + b));
  }
}

enum { MODE_F32 = 0, MODE_BF16 = 1, MODE_CAT_BB = 2, MODE_CAT_BF = 3 };
enum { ACT_ID = 0, ACT_ELU = 1, ACT_SIG = 2 };
enum { DST_BF16 = 0, DST_F32 = 1, DST_SCALAR = 2 };

// Fused 2-layer MLP: Y = act2(elu(X W1 + b1) W2 + b2). 64-row tile, 4 waves.
template <int MODE, int D0, int D1, int HID, int OUT, int ACT2, int DST>
__global__ __launch_bounds__(128) void mlp2_kernel(
    const void* __restrict__ vsrc0, const void* __restrict__ vsrc1,
    const float* __restrict__ w1, const float* __restrict__ b1,
    const float* __restrict__ w2, const float* __restrict__ b2,
    void* __restrict__ vdst, int rows) {
  constexpr int IN   = D0 + D1;
  constexpr int INC  = ((IN + 31) / 32) * 32;
  constexpr int OUTP = ((OUT + 15) / 16) * 16;
  constexpr int NPB  = (HID > OUTP) ? HID : OUTP;
  constexpr int TILE = 64;

  __shared__ unsigned short sA[TILE * 32];   // 4 KB
  __shared__ unsigned short sBt[32 * NPB];   // <= 8 KB
  __shared__ unsigned short sH[TILE * HID];  // 16 KB

  const int tid  = threadIdx.x;
  const int lane = tid & 31;
  const int wave = tid >> 5;
  const int row0 = blockIdx.x * TILE;

  { // ---- layer 1
    v8f acc[HID / 16];
    zero_acc(acc);
    for (int kc = 0; kc < INC; kc += 32) {
      for (int idx = tid; idx < TILE * 32; idx += 128) {
        const int r = idx >> 5, c = idx & 31;
        const int gr = row0 + r, k = kc + c;
        unsigned short v = 0;
        if (gr < rows && k < IN) {
          if (MODE == MODE_F32)
            v = f2bf(((const float*)vsrc0)[(long)gr * IN + k]);
          else if (MODE == MODE_BF16)
            v = ((const unsigned short*)vsrc0)[(long)gr * D0 + k];
          else if (MODE == MODE_CAT_BB)
            v = (k < D0) ? ((const unsigned short*)vsrc0)[(long)gr * D0 + k]
                         : ((const unsigned short*)vsrc1)[(long)gr * D1 + (k - D0)];
          else
            v = (k < D0) ? ((const unsigned short*)vsrc0)[(long)gr * D0 + k]
                         : f2bf(((const float*)vsrc1)[(long)gr * D1 + (k - D0)]);
        }
        sA[idx] = v;
      }
      for (int idx = tid; idx < 32 * HID; idx += 128) {
        const int n = idx >> 5, k = idx & 31;
        float v = (kc + k < IN) ? w1[(long)(kc + k) * HID + n] : 0.f;
        sBt[idx] = f2bf(v);
      }
      __syncthreads();
      const v16bf a = load_a_frag(sA, wave * 16, 0, 32);
#pragma unroll
      for (int t = 0; t < HID / 16; ++t)
        acc[t] = wmma_bf16(a, load_b_frag(sBt, t * 16), acc[t]);
      __syncthreads();
    }
    epi_elu_to_lds<HID>(acc, b1, sH, HID, wave);
    __syncthreads();
  }

  { // ---- layer 2
    v8f acc[OUTP / 16];
    zero_acc(acc);
    lds_gemm<HID, OUTP, OUT, 128>(sH, HID, wave * 16, w2, sBt, acc);
#pragma unroll
    for (int t = 0; t < OUTP / 16; ++t) {
      const int col = t * 16 + (lane & 15);
      const int rb = wave * 16 + ((lane >> 4) << 3);
      const float bb = (col < OUT) ? b2[col] : 0.f;
#pragma unroll
      for (int v = 0; v < 8; ++v) {
        const int gr = row0 + rb + v;
        if (gr >= rows) continue;
        float x = acc[t][v] + bb;
        if (ACT2 == ACT_ELU) x = act_elu(x);
        else if (ACT2 == ACT_SIG) x = act_sig(x);
        if (DST == DST_BF16) {
          if (col < OUT) ((unsigned short*)vdst)[(long)gr * OUT + col] = f2bf(x);
        } else if (DST == DST_F32) {
          if (col < OUT) ((float*)vdst)[(long)gr * OUT + col] = x;
        } else {
          if (col == 0 && t == 0) ((float*)vdst)[gr] = x;
        }
      }
    }
  }
}

// state_enc = sa_effect * sa_att (broadcast over 128 cols)
__global__ __launch_bounds__(256) void scale_kernel(const unsigned short* __restrict__ eff,
                                                    const float* __restrict__ att,
                                                    unsigned short* __restrict__ out, long n) {
  const long i = (long)blockIdx.x * 256 + threadIdx.x;
  if (i < n) out[i] = f2bf(bf2f(eff[i]) * att[i >> 7]);
}

// Fused interaction block on pair-rows r = n*7 + j:
//   pw  = elu MLP_256->256->128([state_enc[n] | state_enc[partner]])
//   eff = elu MLP_128->128->32(pw), att = sigmoid MLP_128->128->1(pw)
//   total_eff[n] += eff * att   (global fp32 atomics)
__global__ __launch_bounds__(96) void interact_kernel(
    const unsigned short* __restrict__ senc,
    const float* __restrict__ w1p, const float* __restrict__ b1p,
    const float* __restrict__ w2p, const float* __restrict__ b2p,
    const float* __restrict__ w1e, const float* __restrict__ b1e,
    const float* __restrict__ w2e, const float* __restrict__ b2e,
    const float* __restrict__ w1a, const float* __restrict__ b1a,
    const float* __restrict__ w2a, const float* __restrict__ b2a,
    float* __restrict__ total_eff, int prows) {
  constexpr int TILE = 48;
  __shared__ unsigned short sBt[32 * 256];   // 16 KB weight chunk
  __shared__ unsigned short sH[TILE * 256];  // 24 KB hidden
  __shared__ unsigned short sPW[TILE * 128]; // 12 KB pairwise output
  unsigned short* sA = sH;                   // alias: pair-gather chunk (phase 1a only)
  float* sAtt = (float*)(sH + TILE * 128);   // alias: upper half of sH (phases 2/3 use ld=128)

  const int tid  = threadIdx.x;
  const int lane = tid & 31;
  const int wave = tid >> 5;
  const int row0 = blockIdx.x * TILE;

  { // ---- pairwise layer 1: 256 -> 256 (gathered concat input)
    v8f acc[16];
    zero_acc(acc);
    for (int kc = 0; kc < 256; kc += 32) {
      for (int idx = tid; idx < TILE * 32; idx += 96) {
        const int r = idx >> 5, c = idx & 31;
        const int gr = row0 + r;
        unsigned short v = 0;
        if (gr < prows) {
          const int k = kc + c;
          const int n = gr / 7, j = gr - n * 7;
          if (k < 128) {
            v = senc[(long)n * 128 + k];
          } else {
            const int i = n & 7, b = n >> 3;
            const int jj = j + (j >= i);
            v = senc[(long)((b << 3) + jj) * 128 + (k - 128)];
          }
        }
        sA[idx] = v;
      }
      for (int idx = tid; idx < 32 * 256; idx += 96) {
        const int n = idx >> 5, k = idx & 31;
        sBt[idx] = f2bf(w1p[(long)(kc + k) * 256 + n]);
      }
      __syncthreads();
      const v16bf a = load_a_frag(sA, wave * 16, 0, 32);
#pragma unroll
      for (int t = 0; t < 16; ++t)
        acc[t] = wmma_bf16(a, load_b_frag(sBt, t * 16), acc[t]);
      __syncthreads();
    }
    epi_elu_to_lds<256>(acc, b1p, sH, 256, wave);
    __syncthreads();
  }
  { // ---- pairwise layer 2: 256 -> 128 -> sPW
    v8f acc[8];
    zero_acc(acc);
    lds_gemm<256, 128, 128, 96>(sH, 256, wave * 16, w2p, sBt, acc);
    epi_elu_to_lds<128>(acc, b2p, sPW, 128, wave);
    __syncthreads();
  }
  { // ---- attention MLP: 128 -> 128 -> 1 (sigmoid), result in sAtt
    v8f acc[8];
    zero_acc(acc);
    lds_gemm<128, 128, 128, 96>(sPW, 128, wave * 16, w1a, sBt, acc);
    epi_elu_to_lds<128>(acc, b1a, sH, 128, wave);
    __syncthreads();
    v8f a2[1];
    zero_acc(a2);
    lds_gemm<128, 16, 1, 96>(sH, 128, wave * 16, w2a, sBt, a2);
    if ((lane & 15) == 0) {
      const int rb = wave * 16 + ((lane >> 4) << 3);
      const float bb = b2a[0];
#pragma unroll
      for (int v = 0; v < 8; ++v) sAtt[rb + v] = act_sig(a2[0][v] + bb);
    }
    __syncthreads();
  }
  { // ---- effect MLP: 128 -> 128 -> 32 (elu), then weighted scatter-add
    v8f acc[8];
    zero_acc(acc);
    lds_gemm<128, 128, 128, 96>(sPW, 128, wave * 16, w1e, sBt, acc);
    epi_elu_to_lds<128>(acc, b1e, sH, 128, wave);
    __syncthreads();
    v8f a2[2];
    zero_acc(a2);
    lds_gemm<128, 32, 32, 96>(sH, 128, wave * 16, w2e, sBt, a2);
#pragma unroll
    for (int t = 0; t < 2; ++t) {
      const int col = t * 16 + (lane & 15);
      const int rb = wave * 16 + ((lane >> 4) << 3);
      const float bb = b2e[col];
#pragma unroll
      for (int v = 0; v < 8; ++v) {
        const int r = rb + v;
        const int gr = row0 + r;
        if (gr < prows) {
          const float x = act_elu(a2[t][v] + bb) * sAtt[r];
          atomicAdd(&total_eff[(long)(gr / 7) * 32 + col], x);
        }
      }
    }
  }
}

extern "C" void kernel_launch(void* const* d_in, const int* in_sizes, int n_in,
                              void* d_out, int out_size, void* d_ws, size_t ws_size,
                              hipStream_t stream) {
  const int N  = in_sizes[0] / 128;  // rows (bs*K)
  const int PR = 7 * N;              // pair rows

  const float* X   = (const float*)d_in[0];
  const float* ACTS = (const float*)d_in[1];
  // params flattened in setup_inputs() dict order, depth-first: w1,b1,w2,b2 per MLP
  // 0 inertia, 1 action_enc, 2 action_effect, 3 action_att, 4 pairwise,
  // 5 int_effect, 6 int_att, 7 final, 8 det_out, 9 lambdas1, 10 lambdas2
  auto P = [&](int m, int t) { return (const float*)d_in[2 + 4 * m + t]; };

  char* wp = (char*)d_ws;
  auto alloc = [&](size_t bytes) { void* q = wp; wp += (bytes + 255) & ~(size_t)255; return q; };
  unsigned short* st_enc_f = (unsigned short*)alloc((size_t)N * 128 * 2);
  unsigned short* a_enc    = (unsigned short*)alloc((size_t)N * 32 * 2);
  unsigned short* sa_eff   = (unsigned short*)alloc((size_t)N * 128 * 2);
  float*          sa_att   = (float*)alloc((size_t)N * 4);
  unsigned short* st_enc   = (unsigned short*)alloc((size_t)N * 128 * 2);
  float*          tot_eff  = (float*)alloc((size_t)N * 32 * 4);
  unsigned short* aggr     = (unsigned short*)alloc((size_t)N * 128 * 2);

  const int gb = (N + 63) / 64;

  // inertia: 128 -> 128 -> 128, elu
  mlp2_kernel<MODE_F32, 128, 0, 128, 128, ACT_ELU, DST_BF16>
      <<<gb, 128, 0, stream>>>(X, nullptr, P(0,0), P(0,1), P(0,2), P(0,3), st_enc_f, N);
  // action_enc: 6 -> 128 -> 32, elu
  mlp2_kernel<MODE_F32, 6, 0, 128, 32, ACT_ELU, DST_BF16>
      <<<gb, 128, 0, stream>>>(ACTS, nullptr, P(1,0), P(1,1), P(1,2), P(1,3), a_enc, N);
  // action_effect: [160] -> 128 -> 128, elu
  mlp2_kernel<MODE_CAT_BB, 128, 32, 128, 128, ACT_ELU, DST_BF16>
      <<<gb, 128, 0, stream>>>(st_enc_f, a_enc, P(2,0), P(2,1), P(2,2), P(2,3), sa_eff, N);
  // action_att: [160] -> 128 -> 1, sigmoid
  mlp2_kernel<MODE_CAT_BB, 128, 32, 128, 1, ACT_SIG, DST_SCALAR>
      <<<gb, 128, 0, stream>>>(st_enc_f, a_enc, P(3,0), P(3,1), P(3,2), P(3,3), sa_att, N);
  // state_enc = sa_effect * sa_att
  {
    long total = (long)N * 128;
    scale_kernel<<<(int)((total + 255) / 256), 256, 0, stream>>>(sa_eff, sa_att, st_enc, total);
  }
  hipMemsetAsync(tot_eff, 0, (size_t)N * 32 * 4, stream);
  // fused pairwise + int_effect + int_att + weighted sum
  interact_kernel<<<(PR + 47) / 48, 96, 0, stream>>>(
      st_enc,
      P(4,0), P(4,1), P(4,2), P(4,3),
      P(5,0), P(5,1), P(5,2), P(5,3),
      P(6,0), P(6,1), P(6,2), P(6,3),
      tot_eff, PR);
  // final: [128 | 32] -> 128 -> 128, elu
  mlp2_kernel<MODE_CAT_BF, 128, 32, 128, 128, ACT_ELU, DST_BF16>
      <<<gb, 128, 0, stream>>>(st_enc, tot_eff, P(7,0), P(7,1), P(7,2), P(7,3), aggr, N);
  // heads: 128 -> 128 -> 64, identity, fp32 to d_out
  float* out = (float*)d_out;
  mlp2_kernel<MODE_BF16, 128, 0, 128, 64, ACT_ID, DST_F32>
      <<<gb, 128, 0, stream>>>(aggr, nullptr, P(8,0), P(8,1), P(8,2), P(8,3), out, N);
  mlp2_kernel<MODE_BF16, 128, 0, 128, 64, ACT_ID, DST_F32>
      <<<gb, 128, 0, stream>>>(aggr, nullptr, P(9,0), P(9,1), P(9,2), P(9,3), out + (size_t)N * 64, N);
  mlp2_kernel<MODE_BF16, 128, 0, 128, 64, ACT_ID, DST_F32>
      <<<gb, 128, 0, stream>>>(aggr, nullptr, P(10,0), P(10,1), P(10,2), P(10,3), out + (size_t)2 * N * 64, N);
}
